// MODEL_90280212561928
// MI455X (gfx1250) — compile-verified, matrix operand order checked by reading, not run on hardware
//
#include <hip/hip_runtime.h>
#include <hip/hip_bf16.h>

typedef __attribute__((ext_vector_type(16))) __bf16 v16bf;
typedef __attribute__((ext_vector_type(8)))  float  v8f;

#define DRUG_NUM 60000
#define DIS_NUM  40000
#define NN       100000       // N_NODES
#define D        128
#define S        8            // HOPS+1
#define NE       800000
#define NLAY     2
#define FD       512
#define MTOK     (NN*S)       // 800000 token rows

__device__ inline float wred(float x) {
#pragma unroll
    for (int m = 16; m; m >>= 1) x += __shfl_xor(x, m, 32);
    return x;
}

// ---------------- weight swizzle: f32 [K][N] row-major -> bf16 WMMA B-fragments
// frag element index = ((kstep*ntiles + ntile)*32 + lane)*16 + j
// lane = khalf*16 + (n%16); j<8 -> k = ks*32 + khalf*8 + j ; j>=8 -> k = ks*32 + 16 + khalf*8 + (j-8)
__global__ void swizzle_w(const float* __restrict__ src, __bf16* __restrict__ dst,
                          int K, int N) {
    int idx = blockIdx.x * 256 + threadIdx.x;
    if (idx >= K * N) return;
    int j     = idx & 15;
    int lane  = (idx >> 4) & 31;
    int t     = idx >> 9;
    int ntiles = N >> 4;
    int ntile = t % ntiles;
    int kstep = t / ntiles;
    int n     = ntile * 16 + (lane & 15);
    int khalf = lane >> 4;
    int g = j >> 3, jj = j & 7;
    int k = kstep * 32 + g * 16 + khalf * 8 + jj;
    dst[idx] = (__bf16)src[(size_t)k * N + n];
}

// ---------------- fill nf[:,0,:] with concatenated embeddings (nf pre-zeroed)
__global__ void init_nf(float* __restrict__ nf, const float* __restrict__ drug,
                        const float* __restrict__ dis) {
    int gid = blockIdx.x * 256 + threadIdx.x;       // NN*32 threads, float4 each
    int n = gid >> 5, c = (gid & 31) * 4;
    if (n >= NN) return;
    float4 v = (n < DRUG_NUM) ? *(const float4*)(drug + (size_t)n * D + c)
                              : *(const float4*)(dis + (size_t)(n - DRUG_NUM) * D + c);
    *(float4*)(nf + (size_t)n * S * D + c) = v;
}

// ---------------- one SpMM hop: nf[:,s,:] += scatter(nf[src, s-1, :] * val)
__global__ void hop_kernel(float* __restrict__ nf, const int* __restrict__ esrc,
                           const int* __restrict__ edst, const float* __restrict__ eval,
                           int s) {
    int gid = blockIdx.x * 256 + threadIdx.x;       // NE*32 threads
    int e = gid >> 5, lane = gid & 31;
    if (e >= NE) return;
    int src = esrc[e], dst = edst[e];
    float w = eval[e];
    const float* sp = nf + (size_t)src * S * D + (size_t)(s - 1) * D + lane * 4;
    float*       dp = nf + (size_t)dst * S * D + (size_t)s * D + lane * 4;
    float4 v = *(const float4*)sp;
    atomicAdd(dp + 0, v.x * w);
    atomicAdd(dp + 1, v.y * w);
    atomicAdd(dp + 2, v.z * w);
    atomicAdd(dp + 3, v.w * w);
}

// ---------------- LayerNorm over last dim (128), wave per row
__global__ void ln_kernel(const float* __restrict__ in, float* __restrict__ out,
                          const float* __restrict__ sc, const float* __restrict__ bb) {
    int w = (blockIdx.x * 256 + threadIdx.x) >> 5;  // row id, MTOK rows
    int lane = threadIdx.x & 31;
    if (w >= MTOK) return;
    int c = lane * 4;
    const float* rp = in + (size_t)w * D + c;
    float4 x = *(const float4*)rp;
    float s1 = wred(x.x + x.y + x.z + x.w) * (1.0f / D);
    float s2 = wred(x.x * x.x + x.y * x.y + x.z * x.z + x.w * x.w) * (1.0f / D);
    float rs = rsqrtf(s2 - s1 * s1 + 1e-5f);
    float4 g = *(const float4*)(sc + c);
    float4 b = *(const float4*)(bb + c);
    float4 o;
    o.x = (x.x - s1) * rs * g.x + b.x;
    o.y = (x.y - s1) * rs * g.y + b.y;
    o.z = (x.z - s1) * rs * g.z + b.z;
    o.w = (x.w - s1) * rs * g.w + b.w;
    *(float4*)(out + (size_t)w * D + c) = o;
}

// ---------------- WMMA GEMM: C[M x N] = A[M x K](f32->bf16) @ Wfrag + epilogue
// epi: 0 = +bias, 1 = gelu(x+bias), 2 = resid + x + bias
__global__ __launch_bounds__(256) void wmma_gemm(
    const float* __restrict__ A, const __bf16* __restrict__ Wfrag,
    const float* __restrict__ bias, const float* __restrict__ resid,
    float* __restrict__ C, int Kdim, int Ndim, int epi) {
    const int wave  = threadIdx.x >> 5;
    const int lane  = threadIdx.x & 31;
    const int m0    = blockIdx.x * 128 + wave * 16;
    const int n0    = blockIdx.y * 128;
    const int arow  = m0 + (lane & 15);
    const int khalf = lane >> 4;
    const int ksteps = Kdim >> 5;
    const int ntiles_total = Ndim >> 4;

    v8f acc[8];
#pragma unroll
    for (int i = 0; i < 8; i++) acc[i] = (v8f)0.0f;

    for (int ks = 0; ks < ksteps; ks++) {
        // A fragment: 16 bf16 per lane from two 8-float runs
        const float* ap = A + (size_t)arow * Kdim + ks * 32 + khalf * 8;
        v16bf a;
#pragma unroll
        for (int g = 0; g < 2; g++) {
            float4 f0 = *(const float4*)(ap + g * 16);
            float4 f1 = *(const float4*)(ap + g * 16 + 4);
            a[g * 8 + 0] = (__bf16)f0.x; a[g * 8 + 1] = (__bf16)f0.y;
            a[g * 8 + 2] = (__bf16)f0.z; a[g * 8 + 3] = (__bf16)f0.w;
            a[g * 8 + 4] = (__bf16)f1.x; a[g * 8 + 5] = (__bf16)f1.y;
            a[g * 8 + 6] = (__bf16)f1.z; a[g * 8 + 7] = (__bf16)f1.w;
        }
#pragma unroll
        for (int nt = 0; nt < 8; nt++) {
            int ntile = (n0 >> 4) + nt;
            v16bf b = *(const v16bf*)(Wfrag +
                        ((size_t)(ks * ntiles_total + ntile) * 32 + lane) * 16);
            acc[nt] = __builtin_amdgcn_wmma_f32_16x16x32_bf16(
                false, a, false, b, (short)0, acc[nt], false, false);
        }
    }
    // epilogue: VGPR r of acc holds row m0 + r + 8*khalf, col n0 + nt*16 + (lane&15)
#pragma unroll
    for (int nt = 0; nt < 8; nt++) {
        int col = n0 + nt * 16 + (lane & 15);
        float bv = bias[col];
#pragma unroll
        for (int r = 0; r < 8; r++) {
            int orow = m0 + r + 8 * khalf;
            size_t oi = (size_t)orow * Ndim + col;
            float v = acc[nt][r] + bv;
            if (epi == 1) v = 0.5f * v * (1.0f + erff(v * 0.70710678118654752f));
            if (epi == 2) v += resid[oi];
            C[oi] = v;
        }
    }
}

// ---------------- per-(node,head) attention over S=8 tokens, dh=32 (lane = d)
__global__ void attn_kernel(const float* __restrict__ Q, const float* __restrict__ K,
                            const float* __restrict__ V, float* __restrict__ O) {
    int w = (blockIdx.x * 256 + threadIdx.x) >> 5;  // NN*4 waves
    int lane = threadIdx.x & 31;
    int node = w >> 2, head = w & 3;
    if (node >= NN) return;
    size_t base = (size_t)node * S * D + head * 32 + lane;
    float q[S], k[S], v[S];
#pragma unroll
    for (int s = 0; s < S; s++) {
        q[s] = Q[base + s * D];
        k[s] = K[base + s * D];
        v[s] = V[base + s * D];
    }
    const float scale = 0.17677669529663687f;       // 32^-0.5
#pragma unroll
    for (int s = 0; s < S; s++) {
        float sc[S], mx = -3.0e38f;
#pragma unroll
        for (int t = 0; t < S; t++) {
            sc[t] = wred(q[s] * k[t]) * scale;
            mx = fmaxf(mx, sc[t]);
        }
        float den = 0.0f, acc = 0.0f;
#pragma unroll
        for (int t = 0; t < S; t++) {
            sc[t] = __expf(sc[t] - mx);
            den += sc[t];
            acc += sc[t] * v[t];
        }
        O[base + s * D] = acc / den;
    }
}

// ---------------- final LN + hop-attention pooling, wave per node
__global__ void pool_kernel(const float* __restrict__ h, const float* __restrict__ fs,
                            const float* __restrict__ fb, const float* __restrict__ aw,
                            const float* __restrict__ ab, float* __restrict__ out) {
    int w = (blockIdx.x * 256 + threadIdx.x) >> 5;  // NN waves
    int lane = threadIdx.x & 31;
    if (w >= NN) return;
    int c = lane * 4;
    float4 g = *(const float4*)(fs + c);
    float4 b = *(const float4*)(fb + c);
    float4 r[S];
#pragma unroll
    for (int s = 0; s < S; s++) {
        float4 x = *(const float4*)(h + (size_t)(w * S + s) * D + c);
        float s1 = wred(x.x + x.y + x.z + x.w) * (1.0f / D);
        float s2 = wred(x.x * x.x + x.y * x.y + x.z * x.z + x.w * x.w) * (1.0f / D);
        float rs = rsqrtf(s2 - s1 * s1 + 1e-5f);
        r[s].x = (x.x - s1) * rs * g.x + b.x;
        r[s].y = (x.y - s1) * rs * g.y + b.y;
        r[s].z = (x.z - s1) * rs * g.z + b.z;
        r[s].w = (x.w - s1) * rs * g.w + b.w;
    }
    float4 wt = *(const float4*)(aw + c);           // attn_w[0:128]
    float4 wn = *(const float4*)(aw + D + c);       // attn_w[128:256]
    float tdot = wred(r[0].x * wt.x + r[0].y * wt.y + r[0].z * wt.z + r[0].w * wt.w);
    float e[S - 1], mx = -3.0e38f;
#pragma unroll
    for (int j = 1; j < S; j++) {
        e[j - 1] = tdot + ab[0] +
            wred(r[j].x * wn.x + r[j].y * wn.y + r[j].z * wn.z + r[j].w * wn.w);
        mx = fmaxf(mx, e[j - 1]);
    }
    float den = 0.0f;
#pragma unroll
    for (int j = 0; j < S - 1; j++) { e[j] = __expf(e[j] - mx); den += e[j]; }
    float inv = 1.0f / den;
    float4 f = r[0];
#pragma unroll
    for (int j = 1; j < S; j++) {
        float a = e[j - 1] * inv;
        f.x += a * r[j].x; f.y += a * r[j].y; f.z += a * r[j].z; f.w += a * r[j].w;
    }
    *(float4*)(out + (size_t)w * D + c) = f;
}

extern "C" void kernel_launch(void* const* d_in, const int* in_sizes, int n_in,
                              void* d_out, int out_size, void* d_ws, size_t ws_size,
                              hipStream_t stream) {
    const float* drug_emb = (const float*)d_in[4];
    const float* dis_emb  = (const float*)d_in[5];
    const int*   esrc     = (const int*)d_in[6];
    const int*   edst     = (const int*)d_in[7];
    const float* eval     = (const float*)d_in[8];
    const float* att_w    = (const float*)d_in[9];
    const float* att_b    = (const float*)d_in[10];
    const float* ln1_s    = (const float*)d_in[11];
    const float* ln1_b    = (const float*)d_in[12];
    const float* q_w = (const float*)d_in[13]; const float* q_b = (const float*)d_in[14];
    const float* k_w = (const float*)d_in[15]; const float* k_b = (const float*)d_in[16];
    const float* v_w = (const float*)d_in[17]; const float* v_b = (const float*)d_in[18];
    const float* o_w = (const float*)d_in[19]; const float* o_b = (const float*)d_in[20];
    const float* ln2_s = (const float*)d_in[21]; const float* ln2_b = (const float*)d_in[22];
    const float* f1_w = (const float*)d_in[23]; const float* f1_b = (const float*)d_in[24];
    const float* f2_w = (const float*)d_in[25]; const float* f2_b = (const float*)d_in[26];
    const float* fln_s = (const float*)d_in[27]; const float* fln_b = (const float*)d_in[28];
    const float* attn_w = (const float*)d_in[29]; const float* attn_b = (const float*)d_in[30];

    const size_t SZ = (size_t)MTOK * D;             // 102,400,000 floats
    float* ws  = (float*)d_ws;
    float* h   = ws;
    float* Y   = ws + SZ;
    float* nf  = ws + 2 * SZ;
    float* Qb  = ws + 3 * SZ;
    float* Kb  = ws + 4 * SZ;
    float* Vb  = ws + 5 * SZ;
    float* MID = nf;                                // spans nf..Vb (4*SZ) once hops done
    __bf16* wst = (__bf16*)(ws + 6 * SZ);

    const int WQ = D * D, WF = D * FD;
    __bf16* w_att = wst;
    __bf16* w_q  = w_att + WQ;
    __bf16* w_k  = w_q + 2 * WQ;
    __bf16* w_v  = w_k + 2 * WQ;
    __bf16* w_o  = w_v + 2 * WQ;
    __bf16* w_f1 = w_o + 2 * WQ;
    __bf16* w_f2 = w_f1 + 2 * WF;

    // weight swizzle (tiny, L2-resident afterwards)
    swizzle_w<<<(WQ + 255) / 256, 256, 0, stream>>>(att_w, w_att, D, D);
    for (int l = 0; l < NLAY; l++) {
        swizzle_w<<<(WQ + 255) / 256, 256, 0, stream>>>(q_w + (size_t)l * WQ, w_q + l * WQ, D, D);
        swizzle_w<<<(WQ + 255) / 256, 256, 0, stream>>>(k_w + (size_t)l * WQ, w_k + l * WQ, D, D);
        swizzle_w<<<(WQ + 255) / 256, 256, 0, stream>>>(v_w + (size_t)l * WQ, w_v + l * WQ, D, D);
        swizzle_w<<<(WQ + 255) / 256, 256, 0, stream>>>(o_w + (size_t)l * WQ, w_o + l * WQ, D, D);
        swizzle_w<<<(WF + 255) / 256, 256, 0, stream>>>(f1_w + (size_t)l * WF, w_f1 + l * WF, D, FD);
        swizzle_w<<<(WF + 255) / 256, 256, 0, stream>>>(f2_w + (size_t)l * WF, w_f2 + l * WF, FD, D);
    }

    // hop features
    hipMemsetAsync(nf, 0, SZ * sizeof(float), stream);
    init_nf<<<(NN * 32) / 256, 256, 0, stream>>>(nf, drug_emb, dis_emb);
    for (int s = 1; s < S; s++)
        hop_kernel<<<(NE * 32) / 256, 256, 0, stream>>>(nf, esrc, edst, eval, s);

    const int GX = MTOK / 128;                      // 6250
    // h = nf @ att_w + att_b
    wmma_gemm<<<dim3(GX, 1), 256, 0, stream>>>(nf, w_att, att_b, nullptr, h, D, D, 0);

    for (int l = 0; l < NLAY; l++) {
        ln_kernel<<<MTOK / 8, 256, 0, stream>>>(h, Y, ln1_s + l * D, ln1_b + l * D);
        wmma_gemm<<<dim3(GX, 1), 256, 0, stream>>>(Y, w_q + l * WQ, q_b + l * D, nullptr, Qb, D, D, 0);
        wmma_gemm<<<dim3(GX, 1), 256, 0, stream>>>(Y, w_k + l * WQ, k_b + l * D, nullptr, Kb, D, D, 0);
        wmma_gemm<<<dim3(GX, 1), 256, 0, stream>>>(Y, w_v + l * WQ, v_b + l * D, nullptr, Vb, D, D, 0);
        attn_kernel<<<(NN * 4) / 8, 256, 0, stream>>>(Qb, Kb, Vb, Y);
        wmma_gemm<<<dim3(GX, 1), 256, 0, stream>>>(Y, w_o + l * WQ, o_b + l * D, h, h, D, D, 2);
        ln_kernel<<<MTOK / 8, 256, 0, stream>>>(h, Y, ln2_s + l * D, ln2_b + l * D);
        wmma_gemm<<<dim3(GX, 4), 256, 0, stream>>>(Y, w_f1 + l * WF, f1_b + l * FD, nullptr, MID, D, FD, 1);
        wmma_gemm<<<dim3(GX, 1), 256, 0, stream>>>(MID, w_f2 + l * WF, f2_b + l * D, h, h, FD, D, 2);
    }

    pool_kernel<<<NN / 8, 256, 0, stream>>>(h, fln_s, fln_b, attn_w, attn_b, (float*)d_out);
}
